// MHABlock_38611755991372
// MI455X (gfx1250) — compile-verified, matrix operand order checked by reading, not run on hardware
//
#include <hip/hip_runtime.h>
#include <hip/hip_bf16.h>
#include <math.h>

#define B_   2
#define N_   2048
#define D_   1536
#define H_   8
#define DK_  64
#define DV_  192
#define NPF_ 192
#define NB_  32
#define M2_  4095   // 2N-1

typedef __attribute__((ext_vector_type(16))) _Float16 v16h;
typedef __attribute__((ext_vector_type(8)))  _Float16 v8h;
typedef __attribute__((ext_vector_type(8)))  float    v8f;
typedef unsigned int v4u __attribute__((ext_vector_type(4)));
typedef int v8i __attribute__((ext_vector_type(8)));
typedef int v4i __attribute__((ext_vector_type(4)));

#if defined(__HIP_DEVICE_COMPILE__) && __has_builtin(__builtin_amdgcn_tensor_load_to_lds)
#define USE_TDM 1
#else
#define USE_TDM 0
#endif

// k index (within a k=32 step) held by element e (0..15) of an A/B fragment, wave32.
// Lanes 0-15: e0..7 -> k0..7, e8..15 -> k16..23 ; lanes 16-31: +8.
__device__ __forceinline__ int frag_k(int lane, int e) {
    return (e & 7) + ((e >> 3) << 4) + ((lane >> 4) << 3);
}

// Load an A fragment row (16 halves with the gfx1250 k-swizzle) as two b128 loads.
__device__ __forceinline__ v16h load_afrag(const _Float16* __restrict__ rowk, int lane) {
    int base = (lane >> 4) << 3;
    v8h a0 = *(const v8h*)(rowk + base);
    v8h a1 = *(const v8h*)(rowk + base + 16);
    return __builtin_shufflevector(a0, a1, 0,1,2,3,4,5,6,7,8,9,10,11,12,13,14,15);
}

__device__ __forceinline__ v16h join8(v8h lo, v8h hi) {
    return __builtin_shufflevector(lo, hi, 0,1,2,3,4,5,6,7,8,9,10,11,12,13,14,15);
}

__device__ __forceinline__ void wait_tensor0() {
#if defined(__HIP_DEVICE_COMPILE__)
#if __has_builtin(__builtin_amdgcn_s_wait_tensorcnt)
    __builtin_amdgcn_s_wait_tensorcnt(0);
#else
    asm volatile("s_wait_tensorcnt 0x0" ::: "memory");
#endif
#endif
}

#if USE_TDM
// 1-D TDM copy: n f16 elements from global to LDS byte-offset lds_off.
// D# group0: count=1 | lds_addr | global_addr | type=2 ; group1: data_size=2B,
// tensor_dim0=tile_dim0=n, tensor_dim1=1, stride0=n. Groups 2/3 unused (<=2D).
__device__ __forceinline__ void tdm_load_1d(unsigned lds_off, const _Float16* gsrc, unsigned n) {
    unsigned long long ga = (unsigned long long)(size_t)gsrc;
    v4u g0 = { 1u,
               lds_off,
               (unsigned)(ga & 0xffffffffu),
               (unsigned)((ga >> 32) & 0x01ffffffu) | (2u << 30) };
    v8i g1 = { (int)(1u << 16),                    // data_size = 1 -> 2 bytes
               (int)((n & 0xffffu) << 16),         // tensor_dim0[15:0]
               (int)((n >> 16) | (1u << 16)),      // tensor_dim0[31:16] | tensor_dim1=1
               (int)((n & 0xffffu) << 16),         // tile_dim0
               0,                                  // tile_dim1/2 = 0
               (int)n,                             // tensor_dim0_stride[31:0]
               0, 0 };
    v4i gz = { 0, 0, 0, 0 };
#if __clang_major__ >= 23
    v8i gz8 = { 0, 0, 0, 0, 0, 0, 0, 0 };
    __builtin_amdgcn_tensor_load_to_lds(g0, g1, gz, gz, gz8, 0);
#else
    __builtin_amdgcn_tensor_load_to_lds(g0, g1, gz, gz, 0);
#endif
}
#endif

// ---------------------------------------------------------------- LayerNorm -> f16
__global__ void ln_cast_kernel(const float* __restrict__ x, const float* __restrict__ g,
                               const float* __restrict__ bta, _Float16* __restrict__ xn) {
    int row = blockIdx.x;                       // 0..B*N-1
    const float* xr = x + (size_t)row * D_;
    __shared__ float red[256];
    float lv[6];
    float s = 0.f;
    #pragma unroll
    for (int i = 0; i < 6; ++i) { lv[i] = xr[threadIdx.x + 256 * i]; s += lv[i]; }
    red[threadIdx.x] = s; __syncthreads();
    for (int o = 128; o > 0; o >>= 1) {
        if (threadIdx.x < o) red[threadIdx.x] += red[threadIdx.x + o];
        __syncthreads();
    }
    float mu = red[0] * (1.f / D_);
    __syncthreads();
    float s2 = 0.f;
    #pragma unroll
    for (int i = 0; i < 6; ++i) { float d = lv[i] - mu; s2 += d * d; }
    red[threadIdx.x] = s2; __syncthreads();
    for (int o = 128; o > 0; o >>= 1) {
        if (threadIdx.x < o) red[threadIdx.x] += red[threadIdx.x + o];
        __syncthreads();
    }
    float rstd = rsqrtf(red[0] * (1.f / D_) + 1e-3f);
    #pragma unroll
    for (int i = 0; i < 6; ++i) {
        int c = threadIdx.x + 256 * i;
        xn[(size_t)row * D_ + c] = (_Float16)((lv[i] - mu) * rstd * g[c] + bta[c]);
    }
}

// ---------------------------------------------------------------- positional embed (2N-1, NPF)
__global__ void pos_embed_kernel(float* __restrict__ pos) {
    int m = blockIdx.x;            // 0..2N-2
    int j = threadIdx.x;           // 0..31 (one wave)
    float dist = (float)(m - (N_ - 1));
    float ad = fabsf(dist);
    float half_life = exp2f(3.f + 8.f * (float)j / (float)(NB_ - 1));
    float e_exp = exp2f(-ad / half_life);
    float cw = exp2f((float)(j + 1)) - 1.f;
    float e_cm = (cw > ad) ? 1.f : 0.f;
    const float stddev = (float)N_ / (2.f * NB_);
    const float start_mean = (float)N_ / NB_;
    float mean = start_mean + ((float)N_ - start_mean) * (float)j / (float)(NB_ - 1);
    float conc = (mean / stddev) * (mean / stddev);
    float rate = mean / (stddev * stddev);
    float log_unnorm = (conc - 1.f) * logf(ad) - rate * ad;
    float log_norm = lgammaf(conc) - conc * logf(rate);
    float prob = expf(log_unnorm - log_norm) + 1e-8f;
    float pm = prob;
    #pragma unroll
    for (int msk = 16; msk; msk >>= 1) pm = fmaxf(pm, __shfl_xor(pm, msk, 32));
    float e_g = prob / pm;
    float sg = (dist > 0.f) ? 1.f : (dist < 0.f ? -1.f : 0.f);
    float* o = pos + (size_t)m * NPF_;
    o[j]            = e_exp;      o[NB_ + j]     = e_cm;      o[2 * NB_ + j] = e_g;
    o[3 * NB_ + j]  = sg * e_exp; o[4 * NB_ + j] = sg * e_cm; o[5 * NB_ + j] = sg * e_g;
}

// ---------------------------------------------------------------- rel_k = pos @ Wrk  (fp32, small)
__global__ void relk_gemm_kernel(const float* __restrict__ pos, const float* __restrict__ Wrk,
                                 float* __restrict__ relk) {
    long idx = (long)blockIdx.x * 256 + threadIdx.x;
    if (idx >= (long)M2_ * (H_ * DK_)) return;
    int m = (int)(idx / (H_ * DK_)), c = (int)(idx % (H_ * DK_));
    const float* p = pos + (size_t)m * NPF_;
    float acc = 0.f;
    for (int f = 0; f < NPF_; ++f) acc = fmaf(p[f], Wrk[(size_t)f * (H_ * DK_) + c], acc);
    relk[idx] = acc;
}

// ---------------------------------------------------------------- pack weights (KxN f32) -> B-fragment f16
__global__ void pack_w_kernel(const float* __restrict__ W, _Float16* __restrict__ Wp,
                              int K, int Nn) {
    long idx = (long)blockIdx.x * 256 + threadIdx.x;
    if (idx >= (long)K * Nn) return;
    int e    = (int)(idx & 15);
    int lane = (int)((idx >> 4) & 31);
    int nt   = (int)((idx >> 9) % (Nn >> 4));
    int kt   = (int)(idx / ((long)512 * (Nn >> 4)));
    int k = kt * 32 + frag_k(lane, e);
    int n = nt * 16 + (lane & 15);
    Wp[idx] = (_Float16)W[(size_t)k * Nn + n];
}

// ---------------------------------------------------------------- pack K (B*N x 512 f32) -> per-head B fragments
__global__ void pack_k_kernel(const float* __restrict__ Kf, _Float16* __restrict__ Kp) {
    long idx = (long)blockIdx.x * 256 + threadIdx.x;   // total = B*H*(N/16)*2*512 = 2^21
    int e    = (int)(idx & 15);
    int lane = (int)((idx >> 4) & 31);
    int kt   = (int)((idx >> 9) & 1);
    int nt   = (int)((idx >> 10) & (N_ / 16 - 1));
    int bh   = (int)(idx >> 17);
    int h = bh % H_, b = bh / H_;
    int key = nt * 16 + (lane & 15);
    int k = kt * 32 + frag_k(lane, e);
    Kp[idx] = (_Float16)Kf[((size_t)(b * N_ + key)) * (H_ * DK_) + h * DK_ + k];
}

// ---------------------------------------------------------------- pack V (B*N x D f32) -> per-head B fragments (key tile 32)
__global__ void pack_v_kernel(const float* __restrict__ Vf, _Float16* __restrict__ Vp) {
    long idx = (long)blockIdx.x * 256 + threadIdx.x;   // total = B*H*(N/32)*12*512
    int e    = (int)(idx & 15);
    int lane = (int)((idx >> 4) & 31);
    int c    = (int)((idx >> 9) % 12);
    int jt   = (int)((idx / (512L * 12)) % (N_ / 32));
    int bh   = (int)(idx / (512L * 12 * (N_ / 32)));
    int h = bh % H_, b = bh / H_;
    int k = frag_k(lane, e);                 // key index within 32-tile
    Vp[idx] = (_Float16)Vf[((size_t)(b * N_ + jt * 32 + k)) * D_ + h * DV_ + c * 16 + (lane & 15)];
}

// ---------------------------------------------------------------- pack rel_k (4095 x 512 f32) -> per-head row-tile B fragments
__global__ void pack_r_kernel(const float* __restrict__ relk, _Float16* __restrict__ Rp) {
    long idx = (long)blockIdx.x * 256 + threadIdx.x;   // total = H*256*2*512 = 2^21
    int e    = (int)(idx & 15);
    int lane = (int)((idx >> 4) & 31);
    int kt   = (int)((idx >> 9) & 1);
    int rt   = (int)((idx >> 10) & 255);
    int h    = (int)(idx >> 18);
    int rrow = rt * 16 + (lane & 15);
    if (rrow > M2_ - 1) rrow = M2_ - 1;                // clamp (never consumed)
    Rp[idx] = (_Float16)relk[(size_t)rrow * (H_ * DK_) + h * DK_ + kt * 32 + frag_k(lane, e)];
}

// ---------------------------------------------------------------- WMMA GEMM: C(MxN,f32) = A(MxK,f16) @ Bpacked (+bias)
// Block = 4 waves; block tile 128x64. Each wave: 32 rows x 64 cols = 8 accumulators.
// The 4 waves share the 64-col B tile, staged in an LDS double buffer by the
// Tensor Data Mover (each wave DMAs its 1KB quarter of the NEXT k-step while
// the current k-step's 8 WMMAs run), synced with s_wait_tensorcnt + barrier.
__global__ void wmma_gemm_kernel(const _Float16* __restrict__ A, const _Float16* __restrict__ Bp,
                                 float* __restrict__ C, const float* __restrict__ bias,
                                 int M, int Nn, int K) {
    int lane = threadIdx.x & 31;
    int wave = __builtin_amdgcn_readfirstlane(threadIdx.x >> 5);   // uniform (SGPR)
    int ntn = Nn >> 4;
    int nblkn = Nn >> 6;
    int bm = blockIdx.x / nblkn;
    int bn = blockIdx.x % nblkn;
    int nt0 = bn * 4;
    const _Float16* arow0 = A + (size_t)(bm * 128 + wave * 32 + (lane & 15)) * K;
    const _Float16* arow1 = arow0 + (size_t)16 * K;
    const v8f vz = {0.f, 0.f, 0.f, 0.f, 0.f, 0.f, 0.f, 0.f};
    v8f acc0[4], acc1[4];
    #pragma unroll
    for (int f = 0; f < 4; ++f) { acc0[f] = vz; acc1[f] = vz; }
    int kSteps = K >> 5;

#if USE_TDM
    __shared__ v8h Bbuf[2][256];                       // 2 x 4KB double buffer
    tdm_load_1d((unsigned)(size_t)&Bbuf[0][wave * 64],
                Bp + ((size_t)(nt0 + wave)) * 512, 512);
    wait_tensor0();
    __syncthreads();
    for (int kt = 0; kt < kSteps; ++kt) {
        int cur = kt & 1;
        if (kt + 1 < kSteps)
            tdm_load_1d((unsigned)(size_t)&Bbuf[cur ^ 1][wave * 64],
                        Bp + ((size_t)((kt + 1) * ntn + nt0 + wave)) * 512, 512);
        v16h a0 = load_afrag(arow0 + kt * 32, lane);
        v16h a1 = load_afrag(arow1 + kt * 32, lane);
        #pragma unroll
        for (int f = 0; f < 4; ++f) {
            v16h bb = join8(Bbuf[cur][f * 64 + lane * 2], Bbuf[cur][f * 64 + lane * 2 + 1]);
            acc0[f] = __builtin_amdgcn_wmma_f32_16x16x32_f16(false, a0, false, bb, (short)0, acc0[f], false, false);
            acc1[f] = __builtin_amdgcn_wmma_f32_16x16x32_f16(false, a1, false, bb, (short)0, acc1[f], false, false);
        }
        wait_tensor0();
        __syncthreads();
    }
#else
    for (int kt = 0; kt < kSteps; ++kt) {
        v16h a0 = load_afrag(arow0 + kt * 32, lane);
        v16h a1 = load_afrag(arow1 + kt * 32, lane);
        #pragma unroll
        for (int f = 0; f < 4; ++f) {
            v16h bb = *(const v16h*)(Bp + ((size_t)(kt * ntn + nt0 + f)) * 512 + lane * 16);
            acc0[f] = __builtin_amdgcn_wmma_f32_16x16x32_f16(false, a0, false, bb, (short)0, acc0[f], false, false);
            acc1[f] = __builtin_amdgcn_wmma_f32_16x16x32_f16(false, a1, false, bb, (short)0, acc1[f], false, false);
        }
    }
#endif

    int rbase = (lane >> 4) << 3;
    int row0 = bm * 128 + wave * 32 + rbase;
    #pragma unroll
    for (int f = 0; f < 4; ++f) {
        int col = (nt0 + f) * 16 + (lane & 15);
        float bv = bias ? bias[col] : 0.f;
        #pragma unroll
        for (int r = 0; r < 8; ++r) {
            C[(size_t)(row0 + r) * Nn + col]      = acc0[f][r] + bv;
            C[(size_t)(row0 + 16 + r) * Nn + col] = acc1[f][r] + bv;
        }
    }
}

// ---------------------------------------------------------------- fused relative attention (flash style)
// grid = B*H*(N/16), block = 32 (one wave). Output f16 (B*N x D) at the head's column slice.
__global__ void attn_kernel(const float* __restrict__ Qf,
                            const _Float16* __restrict__ Kp,
                            const _Float16* __restrict__ Vp,
                            const _Float16* __restrict__ Rp,
                            const float* __restrict__ rcb,
                            const float* __restrict__ rpb,
                            _Float16* __restrict__ Oh) {
    const int lane = threadIdx.x & 31;
    int bt = blockIdx.x;
    int it = bt % (N_ / 16);
    int h  = (bt / (N_ / 16)) % H_;
    int b  = bt / ((N_ / 16) * H_);
    int i0 = it * 16;
    int bh = b * H_ + h;
    const float scale = 0.125f;                        // DK^-0.5

    __shared__ float    Pl[16][48];                    // rel-logit diagonal band
    __shared__ _Float16 Sst[16][32];                   // probability tile staging

    v16h qcA[2], qrA[2];
    {
        int m = lane & 15;
        const float* qrow = Qf + ((size_t)(b * N_ + i0 + m)) * (H_ * DK_) + h * DK_;
        #pragma unroll
        for (int kt = 0; kt < 2; ++kt) {
            #pragma unroll
            for (int e = 0; e < 16; ++e) {
                int k = kt * 32 + frag_k(lane, e);
                float q = qrow[k] * scale;
                qcA[kt][e] = (_Float16)(q + rcb[h * DK_ + k]);
                qrA[kt][e] = (_Float16)(q + rpb[h * DK_ + k]);
            }
        }
    }

    const v8f vz = {0.f, 0.f, 0.f, 0.f, 0.f, 0.f, 0.f, 0.f};
    v8f O[12];
    #pragma unroll
    for (int c = 0; c < 12; ++c) O[c] = vz;
    float mr[8], lr[8];
    #pragma unroll
    for (int r = 0; r < 8; ++r) { mr[r] = -1e30f; lr[r] = 0.f; }

    const int rbase = (lane >> 4) << 3;
    const int col   = lane & 15;

    for (int jt = 0; jt < N_ / 32; ++jt) {
        int j0 = jt * 32;
        int rt0 = ((N_ - 1) + j0 - i0 - 15) >> 4;
        if (jt + 1 < N_ / 32) {                         // prefetch next key tile
            __builtin_prefetch(Kp + ((size_t)((bh * (N_ / 16) + ((j0 + 32) >> 4)) * 2)) * 512 + lane * 16, 0, 0);
            __builtin_prefetch(Vp + ((size_t)((bh * (N_ / 32) + jt + 1) * 12)) * 512 + lane * 16, 0, 0);
            __builtin_prefetch(Rp + ((size_t)((h * 256 + rt0 + 2) * 2)) * 512 + lane * 16, 0, 0);
        }
        // ---- content logits: two 16x16 column tiles, k = 64
        v8f S0 = vz, S1 = vz;
        #pragma unroll
        for (int kt = 0; kt < 2; ++kt) {
            v16h bk0 = *(const v16h*)(Kp + ((size_t)((bh * (N_ / 16) + (j0 >> 4)) * 2 + kt)) * 512 + lane * 16);
            S0 = __builtin_amdgcn_wmma_f32_16x16x32_f16(false, qcA[kt], false, bk0, (short)0, S0, false, false);
            v16h bk1 = *(const v16h*)(Kp + ((size_t)((bh * (N_ / 16) + (j0 >> 4) + 1) * 2 + kt)) * 512 + lane * 16);
            S1 = __builtin_amdgcn_wmma_f32_16x16x32_f16(false, qcA[kt], false, bk1, (short)0, S1, false, false);
        }
        // ---- rel logits band: rel[i,j] = qr_i . rel_k[(N-1)+j-i]; band start multiple of 16
        #pragma unroll
        for (int tt = 0; tt < 3; ++tt) {
            v8f P = vz;
            #pragma unroll
            for (int kt = 0; kt < 2; ++kt) {
                v16h br = *(const v16h*)(Rp + ((size_t)((h * 256 + rt0 + tt) * 2 + kt)) * 512 + lane * 16);
                P = __builtin_amdgcn_wmma_f32_16x16x32_f16(false, qrA[kt], false, br, (short)0, P, false, false);
            }
            #pragma unroll
            for (int r = 0; r < 8; ++r) Pl[rbase + r][tt * 16 + col] = P[r];
        }
        __syncthreads();
        // ---- add rel (relative_shift as diagonal gather) + online softmax
        float alp[8];
        #pragma unroll
        for (int r = 0; r < 8; ++r) {
            int row = rbase + r;
            S0[r] += Pl[row][col - row + 15];
            S1[r] += Pl[row][col + 16 - row + 15];
            float v = fmaxf(S0[r], S1[r]);
            #pragma unroll
            for (int msk = 8; msk; msk >>= 1) v = fmaxf(v, __shfl_xor(v, msk, 32));
            float mnew = fmaxf(mr[r], v);
            alp[r] = __expf(mr[r] - mnew);
            S0[r] = __expf(S0[r] - mnew);
            S1[r] = __expf(S1[r] - mnew);
            float s = S0[r] + S1[r];
            #pragma unroll
            for (int msk = 8; msk; msk >>= 1) s += __shfl_xor(s, msk, 32);
            lr[r] = lr[r] * alp[r] + s;
            mr[r] = mnew;
            Sst[row][col]      = (_Float16)S0[r];
            Sst[row][col + 16] = (_Float16)S1[r];
        }
        __syncthreads();
        // ---- rescale running output, then O += P(16x32) @ V(32x192)
        #pragma unroll
        for (int c = 0; c < 12; ++c)
            #pragma unroll
            for (int r = 0; r < 8; ++r) O[c][r] *= alp[r];
        v16h ap;
        {
            int m = lane & 15;
            #pragma unroll
            for (int e = 0; e < 16; ++e) ap[e] = Sst[m][frag_k(lane, e)];
        }
        #pragma unroll
        for (int c = 0; c < 12; ++c) {
            v16h bv = *(const v16h*)(Vp + ((size_t)((bh * (N_ / 32) + jt) * 12 + c)) * 512 + lane * 16);
            O[c] = __builtin_amdgcn_wmma_f32_16x16x32_f16(false, ap, false, bv, (short)0, O[c], false, false);
        }
        __syncthreads();
    }
    #pragma unroll
    for (int r = 0; r < 8; ++r) {
        float inv = 1.f / lr[r];
        _Float16* orow = Oh + ((size_t)(b * N_ + i0 + rbase + r)) * D_ + h * DV_;
        #pragma unroll
        for (int c = 0; c < 12; ++c)
            orow[c * 16 + col] = (_Float16)(O[c][r] * inv);
    }
}

// ================================================================ host orchestration
extern "C" void kernel_launch(void* const* d_in, const int* in_sizes, int n_in,
                              void* d_out, int out_size, void* d_ws, size_t ws_size,
                              hipStream_t stream) {
    const float* x    = (const float*)d_in[0];
    const float* ln_g = (const float*)d_in[1];
    const float* ln_b = (const float*)d_in[2];
    const float* Wq   = (const float*)d_in[3];
    const float* Wk   = (const float*)d_in[4];
    const float* Wv   = (const float*)d_in[5];
    const float* Wrk  = (const float*)d_in[6];
    const float* rcb  = (const float*)d_in[7];
    const float* rpb  = (const float*)d_in[8];
    const float* Wo   = (const float*)d_in[9];
    const float* bo   = (const float*)d_in[10];
    float* out = (float*)d_out;

    char* w = (char*)d_ws;
    auto alloc = [&](size_t bytes) -> char* {
        char* p = w;
        w += (bytes + 255) & ~(size_t)255;
        return p;
    };
    const size_t BN = (size_t)B_ * N_;
    _Float16* xn_h   = (_Float16*)alloc(BN * D_ * 2);
    _Float16* Wq_p   = (_Float16*)alloc((size_t)D_ * 512 * 2);
    _Float16* Wk_p   = (_Float16*)alloc((size_t)D_ * 512 * 2);
    _Float16* Wv_p   = (_Float16*)alloc((size_t)D_ * D_ * 2);
    _Float16* Wo_p   = (_Float16*)alloc((size_t)D_ * D_ * 2);
    float*    Qf     = (float*)   alloc(BN * 512 * 4);
    float*    Kf     = (float*)   alloc(BN * 512 * 4);
    float*    Vf     = (float*)   alloc(BN * D_ * 4);
    float*    pos    = (float*)   alloc((size_t)M2_ * NPF_ * 4);
    float*    relk_f = (float*)   alloc((size_t)M2_ * 512 * 4);
    _Float16* Kp     = (_Float16*)alloc((size_t)B_ * H_ * (N_ / 16) * 2 * 512 * 2);
    _Float16* Vp     = (_Float16*)alloc((size_t)B_ * H_ * (N_ / 32) * 12 * 512 * 2);
    _Float16* Rp     = (_Float16*)alloc((size_t)H_ * 256 * 2 * 512 * 2);
    _Float16* attnO  = (_Float16*)alloc(BN * D_ * 2);

    ln_cast_kernel<<<(int)BN, 256, 0, stream>>>(x, ln_g, ln_b, xn_h);
    pos_embed_kernel<<<M2_, 32, 0, stream>>>(pos);
    relk_gemm_kernel<<<(M2_ * 512 + 255) / 256, 256, 0, stream>>>(pos, Wrk, relk_f);
    pack_w_kernel<<<(D_ * 512) / 256, 256, 0, stream>>>(Wq, Wq_p, D_, 512);
    pack_w_kernel<<<(D_ * 512) / 256, 256, 0, stream>>>(Wk, Wk_p, D_, 512);
    pack_w_kernel<<<(D_ * D_) / 256, 256, 0, stream>>>(Wv, Wv_p, D_, D_);
    pack_w_kernel<<<(D_ * D_) / 256, 256, 0, stream>>>(Wo, Wo_p, D_, D_);
    // WMMA GEMMs: grid = (M/128) x (N/64), 4 waves/block
    wmma_gemm_kernel<<<(int)(BN / 128) * (512 / 64), 128, 0, stream>>>(xn_h, Wq_p, Qf, nullptr, (int)BN, 512, D_);
    wmma_gemm_kernel<<<(int)(BN / 128) * (512 / 64), 128, 0, stream>>>(xn_h, Wk_p, Kf, nullptr, (int)BN, 512, D_);
    wmma_gemm_kernel<<<(int)(BN / 128) * (D_ / 64), 128, 0, stream>>>(xn_h, Wv_p, Vf, nullptr, (int)BN, D_, D_);
    pack_k_kernel<<<(B_ * H_ * (N_ / 16) * 2 * 512) / 256, 256, 0, stream>>>(Kf, Kp);
    pack_v_kernel<<<(B_ * H_ * (N_ / 32) * 12 * 512) / 256, 256, 0, stream>>>(Vf, Vp);
    pack_r_kernel<<<(H_ * 256 * 2 * 512) / 256, 256, 0, stream>>>(relk_f, Rp);
    attn_kernel<<<B_ * H_ * (N_ / 16), 32, 0, stream>>>(Qf, Kp, Vp, Rp, rcb, rpb, attnO);
    wmma_gemm_kernel<<<(int)(BN / 128) * (D_ / 64), 128, 0, stream>>>(attnO, Wo_p, out, bo, (int)BN, D_, D_);
}
